// Attn_VarLevel_46385646797319
// MI455X (gfx1250) — matrix operand order repaired and checked by reading, not compiled, not run on hardware
//
#include <hip/hip_runtime.h>
#include <hip/hip_bf16.h>

// ---------------------------------------------------------------------------
// Fused var-level attention for MI455X (gfx1250, wave32, WMMA bf16).
//   B=4, P=512, V=64, D=512.  One workgroup (256 thr = 8 waves) per (b,p).
//   All intermediates (q,k,v,scores,attn,out) stay in LDS; only the three
//   input tensors are read from HBM and the final f32 result written back.
//   Weights are pre-converted to bf16 W^T ([N][K] row-major) in d_ws so that
//   WMMA B-fragments are two contiguous 16B per-lane loads (L2-resident, 1.5MB).
// ---------------------------------------------------------------------------

typedef __attribute__((ext_vector_type(16))) __bf16 bf16x16;
typedef __attribute__((ext_vector_type(8)))  __bf16 bf16x8;
typedef __attribute__((ext_vector_type(8)))  float  f32x8;
typedef __attribute__((ext_vector_type(4)))  float  f32x4;

static constexpr int Bn = 4, Pn = 512, Vn = 64, Dn = 512;

// Padded LDS row strides (elements) to avoid 64-bank conflicts.
static constexpr int QK_STR = 520;  // q / k / out buffers (bf16)
static constexpr int VT_STR = 64;   // v^T buffer (bf16), reuses q buffer
static constexpr int SC_STR = 68;   // scores (f32)
static constexpr int AT_STR = 72;   // attention probabilities (bf16)

__device__ __forceinline__ f32x8 zero8() {
  f32x8 z;
#pragma unroll
  for (int i = 0; i < 8; ++i) z[i] = 0.0f;
  return z;
}

__device__ __forceinline__ f32x8 wmma_bf16(bf16x16 a, bf16x16 b, f32x8 c) {
  // D = A(16x32) * B(32x16) + C, f32 accumulate
  return __builtin_amdgcn_wmma_f32_16x16x32_bf16(
      /*neg_a=*/false, a, /*neg_b=*/false, b,
      /*c_mod=*/(short)0, c, /*reuse_a=*/false, /*reuse_b=*/false);
}

// A fragment (16x32, bf16) from a row-major bf16 matrix (LDS or global).
// ISA layout: lanes 0-15 hold row M=lane, K {k0..k0+7, k0+16..k0+23};
//             lanes 16-31 hold row M=lane-16, K {k0+8..k0+15, k0+24..k0+31}.
__device__ __forceinline__ bf16x16 ldA_bf16(const __bf16* base, int stride,
                                            int m0, int k0, int lane) {
  const int row = m0 + (lane & 15);
  const int kk  = k0 + ((lane >> 4) << 3);
  const __bf16* p = base + row * stride + kk;
  bf16x8 lo = *(const bf16x8*)(p);        // K kk..kk+7    (VGPR 0..3)
  bf16x8 hi = *(const bf16x8*)(p + 16);   // K kk+16..kk+23 (VGPR 4..7)
  bf16x16 r;
#pragma unroll
  for (int i = 0; i < 8; ++i) { r[i] = lo[i]; r[8 + i] = hi[i]; }
  return r;
}

// A fragment from row-major f32 global memory, converted to bf16 on the fly.
__device__ __forceinline__ bf16x16 ldA_f32(const float* base, int stride,
                                           int m0, int k0, int lane) {
  const int row = m0 + (lane & 15);
  const int kk  = k0 + ((lane >> 4) << 3);
  const float* p = base + row * stride + kk;
  f32x4 a0 = *(const f32x4*)(p);
  f32x4 a1 = *(const f32x4*)(p + 4);
  f32x4 a2 = *(const f32x4*)(p + 16);
  f32x4 a3 = *(const f32x4*)(p + 20);
  bf16x16 r;
#pragma unroll
  for (int i = 0; i < 4; ++i) {
    r[i]      = (__bf16)a0[i];
    r[4 + i]  = (__bf16)a1[i];
    r[8 + i]  = (__bf16)a2[i];
    r[12 + i] = (__bf16)a3[i];
  }
  return r;
}

// B fragment (32x16, bf16) from an [N][K] row-major matrix (i.e. B^T stored
// row-major). Lane n = n0+(lane&15), K {kk..kk+7, kk+16..kk+23} with
// kk = k0 + (lane>>4)*8 (same K order as the A fragment).
__device__ __forceinline__ bf16x16 ldB_bf16(const __bf16* base, int stride,
                                            int n0, int k0, int lane) {
  const int col = n0 + (lane & 15);
  const int kk  = k0 + ((lane >> 4) << 3);
  const __bf16* p = base + col * stride + kk;
  bf16x8 lo = *(const bf16x8*)(p);
  bf16x8 hi = *(const bf16x8*)(p + 16);
  bf16x16 r;
#pragma unroll
  for (int i = 0; i < 8; ++i) { r[i] = lo[i]; r[8 + i] = hi[i]; }
  return r;
}

// Store a 16x16 f32 C tile (+bias) as bf16, row-major destination.
__device__ __forceinline__ void stC_row_bf16(__bf16* dst, int stride, int m0,
                                             int n0, int lane, f32x8 acc,
                                             float bias) {
  const int col = n0 + (lane & 15);
  const int r0  = m0 + ((lane >> 4) << 3);
#pragma unroll
  for (int i = 0; i < 8; ++i)
    dst[(r0 + i) * stride + col] = (__bf16)(acc[i] + bias);
}

// Store a 16x16 f32 C tile (+bias) as bf16, TRANSPOSED (dst[n][m]).
// Per lane this is one contiguous 16B bf16x8 store.
__device__ __forceinline__ void stC_tr_bf16(__bf16* dst, int stride, int m0,
                                            int n0, int lane, f32x8 acc,
                                            float bias) {
  const int col = n0 + (lane & 15);
  const int r0  = m0 + ((lane >> 4) << 3);
  bf16x8 v;
#pragma unroll
  for (int i = 0; i < 8; ++i) v[i] = (__bf16)(acc[i] + bias);
  *(bf16x8*)(dst + col * stride + r0) = v;
}

// C[64,512] = A_f32[64,512] @ W (W given as W^T [512][512] bf16) + bias,
// written to LDS as bf16 (row-major or transposed). Wave w owns N columns
// [64w, 64w+64): 4 M-tiles x 4 N-tiles, K = 512 -> 16 WMMA k-steps.
template <bool TRANSPOSED>
__device__ __forceinline__ void projection(const float* __restrict__ A,
                                           const __bf16* __restrict__ WT,
                                           const float* __restrict__ bias,
                                           __bf16* dst, int dstride, int wave,
                                           int lane) {
  const int n0w = wave * 64;
  f32x8 acc[4][4];
#pragma unroll
  for (int mt = 0; mt < 4; ++mt)
#pragma unroll
    for (int nt = 0; nt < 4; ++nt) acc[mt][nt] = zero8();

  for (int ks = 0; ks < 16; ++ks) {
    const int k0 = ks * 32;
    bf16x16 af[4];
#pragma unroll
    for (int mt = 0; mt < 4; ++mt) af[mt] = ldA_f32(A, Dn, mt * 16, k0, lane);
#pragma unroll
    for (int nt = 0; nt < 4; ++nt) {
      bf16x16 bf = ldB_bf16(WT, Dn, n0w + nt * 16, k0, lane);
#pragma unroll
      for (int mt = 0; mt < 4; ++mt)
        acc[mt][nt] = wmma_bf16(af[mt], bf, acc[mt][nt]);
    }
  }
#pragma unroll
  for (int nt = 0; nt < 4; ++nt) {
    const float bv = bias[n0w + nt * 16 + (lane & 15)];
#pragma unroll
    for (int mt = 0; mt < 4; ++mt) {
      if (TRANSPOSED)
        stC_tr_bf16(dst, dstride, mt * 16, n0w + nt * 16, lane, acc[mt][nt], bv);
      else
        stC_row_bf16(dst, dstride, mt * 16, n0w + nt * 16, lane, acc[mt][nt], bv);
    }
  }
}

// ---------------------------------------------------------------------------
// Kernel 0: Wq/Wkv/Wo (f32 [K][N]) -> bf16 W^T ([N][K]) in workspace.
// ---------------------------------------------------------------------------
__global__ __launch_bounds__(256) void prep_weights(
    const float* __restrict__ Wq, const float* __restrict__ Wkv,
    const float* __restrict__ Wo, __bf16* __restrict__ wt) {
  const int g   = blockIdx.x * 256 + threadIdx.x;  // 3 * 512 * 512 threads
  const int mat = g >> 18;
  const int rem = g & ((1 << 18) - 1);
  const int n   = rem >> 9;
  const int k   = rem & 511;
  const float* W = (mat == 0) ? Wq : (mat == 1) ? Wkv : Wo;
  wt[((size_t)mat << 18) + n * 512 + k] = (__bf16)W[k * 512 + n];
}

// ---------------------------------------------------------------------------
// Kernel 1: fused attention, one workgroup per (b,p).
// ---------------------------------------------------------------------------
__global__ __launch_bounds__(256) void attn_fused(
    const float* __restrict__ queries, const float* __restrict__ keys,
    const float* __restrict__ values, const unsigned char* __restrict__ vmask,
    const float* __restrict__ bq, const float* __restrict__ bkv,
    const float* __restrict__ bo, const __bf16* __restrict__ WqT,
    const __bf16* __restrict__ WkvT, const __bf16* __restrict__ WoT,
    float* __restrict__ out) {
  // 159744 B total (<= 160KB in case CU-mode halves the 320KB WGP LDS pool).
  __shared__ __align__(16) unsigned char smem[159744];
  __bf16* bufA = (__bf16*)smem;             // q (64x520), later v^T (512x64)
  __bf16* bufB = (__bf16*)(smem + 66560);   // k (64x520), later out (64x520)
  float*  sc   = (float*)(smem + 133120);   // scores 64x68 f32
  __bf16* attn = (__bf16*)(smem + 150528);  // probs  64x72 bf16

  const int tid  = threadIdx.x;
  const int lane = tid & 31;
  const int wave = tid >> 5;
  const int bp   = blockIdx.x;       // b*P + p
  const int b    = bp >> 9;          // P = 512
  const size_t base = (size_t)bp * (Vn * Dn);

  // --- Phase 1: q and k projections -> LDS (bf16) -------------------------
  projection<false>(queries + base, WqT, bq, bufA, QK_STR, wave, lane);
  projection<false>(keys + base, WkvT, bkv, bufB, QK_STR, wave, lane);
  __syncthreads();

  // --- Phase 2: scores = q @ k^T (k stored [r][d] == [N][K]) --------------
  {
#pragma unroll
    for (int tt = 0; tt < 2; ++tt) {   // 16 tiles / 8 waves = 2 per wave
      const int t  = wave * 2 + tt;
      const int mt = t >> 2, nt = t & 3;
      f32x8 acc = zero8();
      for (int ks = 0; ks < 16; ++ks) {
        bf16x16 af = ldA_bf16(bufA, QK_STR, mt * 16, ks * 32, lane);
        bf16x16 bf = ldB_bf16(bufB, QK_STR, nt * 16, ks * 32, lane);
        acc = wmma_bf16(af, bf, acc);
      }
      const int col = nt * 16 + (lane & 15);
      const int r0  = mt * 16 + ((lane >> 4) << 3);
#pragma unroll
      for (int i = 0; i < 8; ++i) sc[(r0 + i) * SC_STR + col] = acc[i];
    }
  }
  __syncthreads();

  // --- Phase 3: v projection (transposed, reuses q buffer) + softmax ------
  projection<true>(values + base, WkvT, bkv, bufA, VT_STR, wave, lane);
  if (tid < Vn) {  // waves 0..1 entirely: uniform per-wave control flow
    const int row = tid;
    const unsigned char* mrow = vmask + ((size_t)b * Vn + row) * Vn;
    const float scale = 0.04419417382415922f;  // 1/sqrt(512)
    const float ninf  = -__builtin_inff();
    float mx = ninf;
    for (int c = 0; c < Vn; ++c) {
      float s = mrow[c] ? ninf : scale * sc[row * SC_STR + c];
      sc[row * SC_STR + c] = s;
      mx = fmaxf(mx, s);
    }
    float sum = 0.f;
    for (int c = 0; c < Vn; ++c) {
      float e = (mx == ninf) ? 0.f : __expf(sc[row * SC_STR + c] - mx);
      sc[row * SC_STR + c] = e;
      sum += e;
    }
    const float rinv = (sum > 0.f) ? (1.f / sum) : 0.f;
    for (int c = 0; c < Vn; ++c)
      attn[row * AT_STR + c] = (__bf16)(sc[row * SC_STR + c] * rinv);
  }
  __syncthreads();

  // --- Phase 4: out = attn @ v (v^T stored [d][r] == [N][K]) -> bufB ------
  {
    f32x8 acc[4][4];
#pragma unroll
    for (int mt = 0; mt < 4; ++mt)
#pragma unroll
      for (int nt = 0; nt < 4; ++nt) acc[mt][nt] = zero8();
#pragma unroll
    for (int ks = 0; ks < 2; ++ks) {  // K = 64
      const int k0 = ks * 32;
      bf16x16 af[4];
#pragma unroll
      for (int mt = 0; mt < 4; ++mt)
        af[mt] = ldA_bf16(attn, AT_STR, mt * 16, k0, lane);
#pragma unroll
      for (int nt = 0; nt < 4; ++nt) {
        bf16x16 bf = ldB_bf16(bufA, VT_STR, wave * 64 + nt * 16, k0, lane);
#pragma unroll
        for (int mt = 0; mt < 4; ++mt)
          acc[mt][nt] = wmma_bf16(af[mt], bf, acc[mt][nt]);
      }
    }
#pragma unroll
    for (int nt = 0; nt < 4; ++nt)
#pragma unroll
      for (int mt = 0; mt < 4; ++mt)
        stC_row_bf16(bufB, QK_STR, mt * 16, wave * 64 + nt * 16, lane,
                     acc[mt][nt], 0.f);
  }
  __syncthreads();

  // --- Phase 5: final = out @ Wo + bo -> global f32 -----------------------
  {
    const int n0w = wave * 64;
    f32x8 acc[4][4];
#pragma unroll
    for (int mt = 0; mt < 4; ++mt)
#pragma unroll
      for (int nt = 0; nt < 4; ++nt) acc[mt][nt] = zero8();

    for (int ks = 0; ks < 16; ++ks) {
      const int k0 = ks * 32;
      bf16x16 af[4];
#pragma unroll
      for (int mt = 0; mt < 4; ++mt)
        af[mt] = ldA_bf16(bufB, QK_STR, mt * 16, k0, lane);
#pragma unroll
      for (int nt = 0; nt < 4; ++nt) {
        bf16x16 bf = ldB_bf16(WoT, Dn, n0w + nt * 16, k0, lane);
#pragma unroll
        for (int mt = 0; mt < 4; ++mt)
          acc[mt][nt] = wmma_bf16(af[mt], bf, acc[mt][nt]);
      }
    }
    float* obase = out + base;
#pragma unroll
    for (int nt = 0; nt < 4; ++nt) {
      const int col = n0w + nt * 16 + (lane & 15);
      const float bias = bo[col];
#pragma unroll
      for (int mt = 0; mt < 4; ++mt) {
        const int r0 = mt * 16 + ((lane >> 4) << 3);
#pragma unroll
        for (int i = 0; i < 8; ++i)
          obase[(size_t)(r0 + i) * Dn + col] = acc[mt][nt][i] + bias;
      }
    }
  }
}

// ---------------------------------------------------------------------------
extern "C" void kernel_launch(void* const* d_in, const int* in_sizes, int n_in,
                              void* d_out, int out_size, void* d_ws,
                              size_t ws_size, hipStream_t stream) {
  const float* queries = (const float*)d_in[0];
  const float* keys    = (const float*)d_in[1];
  const float* values  = (const float*)d_in[2];
  const unsigned char* vmask = (const unsigned char*)d_in[3];  // jax bool_: 1B
  const float* Wq  = (const float*)d_in[4];
  const float* bq  = (const float*)d_in[5];
  const float* Wkv = (const float*)d_in[6];
  const float* bkv = (const float*)d_in[7];
  const float* Wo  = (const float*)d_in[8];
  const float* bo  = (const float*)d_in[9];

  __bf16* wt = (__bf16*)d_ws;  // 3 * 512*512 bf16 = 1.5 MB of workspace

  prep_weights<<<3 * 512 * 512 / 256, 256, 0, stream>>>(Wq, Wkv, Wo, wt);
  attn_fused<<<Bn * Pn, 256, 0, stream>>>(
      queries, keys, values, vmask, bq, bkv, bo, wt, wt + (1 << 18),
      wt + (2 << 18), (float*)d_out);
}
